// MultiHeadAttention_81655918232272
// MI455X (gfx1250) — matrix operand order repaired and checked by reading, not compiled, not run on hardware
//
#include <hip/hip_runtime.h>
#include <cstdint>

// ---------------------------------------------------------------------------
// Types for CDNA5 WMMA (gfx1250, wave32)
// ---------------------------------------------------------------------------
typedef __attribute__((ext_vector_type(16))) __bf16 bf16x16;
typedef __attribute__((ext_vector_type(8)))  float  f32x8;

union Frag {
    bf16x16 v;
    uint4 u[2];
    unsigned short h[16];
};

// float -> bf16 round-to-nearest-even
__device__ __forceinline__ unsigned short f2bf(float f) {
    unsigned int u = __float_as_uint(f);
    unsigned int r = u + 0x7fffu + ((u >> 16) & 1u);
    return (unsigned short)(r >> 16);
}

// xor-swizzle within 32 lanes via ds_swizzle (group-of-32 mode: and=0x1f, or=0, xor=m)
#define SWZ(x, m) __int_as_float(__builtin_amdgcn_ds_swizzle(__float_as_int(x), (((m) << 10) | 0x1f)))

// CDNA5 async global->LDS copy (GLOBAL_LOAD_ASYNC_TO_LDS_B128, ASYNCcnt-tracked).
// lds_off: LDS byte address (low 32 bits of generic shared pointer)
// gvoff  : per-lane signed 32-bit byte offset (GVS mode)
// sbase  : uniform 64-bit global base address (SGPR pair)
__device__ __forceinline__ void async_copy_b128(unsigned lds_off, unsigned gvoff,
                                                const void* sbase) {
    asm volatile("global_load_async_to_lds_b128 %0, %1, %2"
                 :
                 : "v"(lds_off), "v"(gvoff), "s"(sbase)
                 : "memory");
}

#define WAIT_ASYNC(n) asm volatile("s_wait_asynccnt %0" ::"n"(n) : "memory")

// ---------------------------------------------------------------------------
// Problem constants (from reference)
// ---------------------------------------------------------------------------
#define DIMK   1024
#define HEADS  16
#define HSZ    64

// GEMM tiling
#define BM  128
#define BN  128
#define BK  32
#define LDK 40   // padded LDS row length (ushorts): 32 data + 8 pad -> conflict-free frag reads

// ---------------------------------------------------------------------------
// Kernel 1: f32 -> bf16 bulk convert (x)
// ---------------------------------------------------------------------------
__global__ __launch_bounds__(256) void cvt_f32_to_bf16(
    const float* __restrict__ in, unsigned short* __restrict__ out, long n)
{
    long i = ((long)blockIdx.x * blockDim.x + threadIdx.x) * 4;
    if (i + 3 < n) {
        float4 f = *(const float4*)(in + i);
        unsigned int lo = (unsigned int)f2bf(f.x) | ((unsigned int)f2bf(f.y) << 16);
        unsigned int hi = (unsigned int)f2bf(f.z) | ((unsigned int)f2bf(f.w) << 16);
        uint2 p; p.x = lo; p.y = hi;
        *(uint2*)(out + i) = p;
    }
}

// ---------------------------------------------------------------------------
// Kernel 2: W[k][n] f32 -> WT[n][k] bf16 (tiled transpose through LDS)
// ---------------------------------------------------------------------------
__global__ __launch_bounds__(256) void transpose_cvt(
    const float* __restrict__ W, unsigned short* __restrict__ WT, int D)
{
    __shared__ unsigned short tile[64][72];
    const int bn = blockIdx.x * 64;
    const int bk = blockIdx.y * 64;
#pragma unroll
    for (int i = 0; i < 16; ++i) {
        int lin = threadIdx.x + i * 256;   // 0..4095
        int kk = lin >> 6, nn = lin & 63;
        tile[kk][nn] = f2bf(W[(size_t)(bk + kk) * D + (bn + nn)]);
    }
    __syncthreads();
#pragma unroll
    for (int i = 0; i < 16; ++i) {
        int lin = threadIdx.x + i * 256;
        int nn = lin >> 6, kk = lin & 63;
        WT[(size_t)(bn + nn) * D + (bk + kk)] = tile[kk][nn];
    }
}

// ---------------------------------------------------------------------------
// Kernel 3: bf16 WMMA GEMM   C[M][N] = A[M][K] * BT[N][K]^T + bias
//   A row-major bf16, BT = B-transposed row-major bf16 (so B-frag reads are
//   contiguous), f32 accumulation, output bf16 or f32.
//   Block: 256 threads (8 waves). Block tile 128x128, BK=32.
//   Wave grid 4x2: each wave computes 2x4 tiles of 16x16 -> 8 WMMA / stage.
//   Global->LDS staging uses CDNA5 async-to-LDS copies (ASYNCcnt) with a
//   2-deep software pipeline: stage s+1 copies overlap stage s WMMAs.
// ---------------------------------------------------------------------------
template <bool OUT_BF16>
__global__ __launch_bounds__(256) void wmma_gemm(
    const unsigned short* __restrict__ A,
    const unsigned short* __restrict__ BT,
    const float* __restrict__ bias,
    void* __restrict__ Cout,
    int M, int N, int K)
{
    __shared__ __align__(16) unsigned short As[2][BM * LDK];
    __shared__ __align__(16) unsigned short Bs[2][BN * LDK];

    const int tid  = threadIdx.x;
    const int lane = tid & 31;
    const int wave = tid >> 5;
    const int hi   = lane >> 4;     // half-wave select (ISA frag layout)
    const int lm   = lane & 15;
    const int wr   = wave & 3;      // 4 row groups of 32
    const int wc   = wave >> 2;     // 2 col groups of 64

    const size_t rowBlock = (size_t)blockIdx.y * BM;
    const size_t colBlock = (size_t)blockIdx.x * BN;

    // global->LDS mapping: 512 16B-chunks per tile, 2 per thread per matrix
    const int r0 = tid >> 2;          // 0..63
    const int c0 = (tid & 3) * 8;     // 0,8,16,24

    // uniform global bases (SGPR pairs in the async copies)
    const unsigned short* Abase = A  + rowBlock * (size_t)K;
    const unsigned short* Bbase = BT + colBlock * (size_t)K;

    // per-lane global byte offsets (stage 0)
    const unsigned aoff0 = (unsigned)(((unsigned)r0        * (unsigned)K + (unsigned)c0) * 2u);
    const unsigned aoff1 = (unsigned)((((unsigned)r0 + 64u) * (unsigned)K + (unsigned)c0) * 2u);

    // per-lane LDS byte addresses for both buffers
    unsigned ldsA0[2], ldsA1[2], ldsB0[2], ldsB1[2];
#pragma unroll
    for (int b = 0; b < 2; ++b) {
        ldsA0[b] = (unsigned)(uintptr_t)&As[b][r0 * LDK + c0];
        ldsA1[b] = (unsigned)(uintptr_t)&As[b][(r0 + 64) * LDK + c0];
        ldsB0[b] = (unsigned)(uintptr_t)&Bs[b][r0 * LDK + c0];
        ldsB1[b] = (unsigned)(uintptr_t)&Bs[b][(r0 + 64) * LDK + c0];
    }

    f32x8 acc[2][4];
#pragma unroll
    for (int mt = 0; mt < 2; ++mt)
#pragma unroll
        for (int nt = 0; nt < 4; ++nt)
#pragma unroll
            for (int e = 0; e < 8; ++e) acc[mt][nt][e] = 0.0f;

    const int stages = K / BK;

    // prologue: async-issue stage 0 into buffer 0
    {
        async_copy_b128(ldsA0[0], aoff0, Abase);
        async_copy_b128(ldsA1[0], aoff1, Abase);
        async_copy_b128(ldsB0[0], aoff0, Bbase);
        async_copy_b128(ldsB1[0], aoff1, Bbase);
    }

    for (int s = 0; s < stages; ++s) {
        const int cur = s & 1;
        const int nxt = cur ^ 1;

        if (s + 1 < stages) {
            // issue stage s+1 copies; they overlap this stage's WMMAs
            const unsigned gs = (unsigned)((s + 1) * BK * 2);
            async_copy_b128(ldsA0[nxt], aoff0 + gs, Abase);
            async_copy_b128(ldsA1[nxt], aoff1 + gs, Abase);
            async_copy_b128(ldsB0[nxt], aoff0 + gs, Bbase);
            async_copy_b128(ldsB1[nxt], aoff1 + gs, Bbase);
            WAIT_ASYNC(4);   // retire stage-s copies (async ops complete in order)
        } else {
            WAIT_ASYNC(0);
        }
        __syncthreads();

        Frag af[2], bfr[4];
#pragma unroll
        for (int mt = 0; mt < 2; ++mt) {
            const unsigned short* base = &As[cur][(wr * 32 + mt * 16 + lm) * LDK];
            // A frag: lane<16: K0..7 & K16..23 ; lane>=16: K8..15 & K24..31
            af[mt].u[0] = *(const uint4*)(base + hi * 8);
            af[mt].u[1] = *(const uint4*)(base + 16 + hi * 8);
        }
#pragma unroll
        for (int nt = 0; nt < 4; ++nt) {
            const unsigned short* base = &Bs[cur][(wc * 64 + nt * 16 + lm) * LDK];
            // B frag: lane = column, lane<16: K0..7 & K16..23 ; lane>=16: K8..15 & K24..31
            bfr[nt].u[0] = *(const uint4*)(base + hi * 8);
            bfr[nt].u[1] = *(const uint4*)(base + 16 + hi * 8);
        }
#pragma unroll
        for (int mt = 0; mt < 2; ++mt)
#pragma unroll
            for (int nt = 0; nt < 4; ++nt)
                acc[mt][nt] = __builtin_amdgcn_wmma_f32_16x16x32_bf16(
                    false, af[mt].v, false, bfr[nt].v, (short)0, acc[mt][nt], false, false);

        // all waves finished reading `cur` before anyone re-issues writes into it
        __syncthreads();
    }

    // epilogue: C layout -> lane<16: M=r, N=lane ; lane>=16: M=8+r, N=lane-16
#pragma unroll
    for (int mt = 0; mt < 2; ++mt) {
        const size_t growBase = rowBlock + wr * 32 + mt * 16 + 8 * hi;
#pragma unroll
        for (int nt = 0; nt < 4; ++nt) {
            const size_t gcol = colBlock + wc * 64 + nt * 16 + lm;
            const float b = bias ? bias[gcol] : 0.0f;
#pragma unroll
            for (int r = 0; r < 8; ++r) {
                const float val = acc[mt][nt][r] + b;
                const size_t off = (growBase + r) * (size_t)N + gcol;
                if (OUT_BF16) ((unsigned short*)Cout)[off] = f2bf(val);
                else          ((float*)Cout)[off] = val;
            }
        }
    }
}

// ---------------------------------------------------------------------------
// Kernel 4: per-token attention (HEADS x HEADS scores across heads, softmax,
// ctx). One wave per token; scores + ctx via WMMA; softmax with ds_swizzle
// xor-reductions inside each 16-lane half (matches C-matrix lane layout).
// ---------------------------------------------------------------------------
__global__ __launch_bounds__(256) void attention_kernel(
    const unsigned short* __restrict__ Q,
    const unsigned short* __restrict__ Kb,
    const unsigned short* __restrict__ Vb,
    unsigned short* __restrict__ Ctx)
{
    __shared__ __align__(16) unsigned short attnS[8][16][16];

    const int tid  = threadIdx.x;
    const int lane = tid & 31;
    const int wave = tid >> 5;
    const int hi   = lane >> 4;
    const int lm   = lane & 15;

    const size_t token = (size_t)blockIdx.x * 8 + wave;
    const unsigned short* q = Q  + token * (size_t)DIMK;   // [16 heads][64 dims]
    const unsigned short* k = Kb + token * (size_t)DIMK;
    const unsigned short* v = Vb + token * (size_t)DIMK;

    // ---- scores = Q(16x64) @ K^T(64x16), two K-depth-32 WMMAs ----
    f32x8 s;
#pragma unroll
    for (int e = 0; e < 8; ++e) s[e] = 0.0f;
#pragma unroll
    for (int j = 0; j < 2; ++j) {
        Frag qa, kf;
        const unsigned short* qrow = q + lm * HSZ + j * 32;     // A: row = head
        qa.u[0] = *(const uint4*)(qrow + hi * 8);
        qa.u[1] = *(const uint4*)(qrow + 16 + hi * 8);
        const unsigned short* krow = k + lm * HSZ + j * 32 + hi * 8;  // B col = head g
        kf.u[0] = *(const uint4*)krow;
        kf.u[1] = *(const uint4*)(krow + 16);
        s = __builtin_amdgcn_wmma_f32_16x16x32_bf16(
            false, qa.v, false, kf.v, (short)0, s, false, false);
    }

    // ---- row softmax over g (16 lanes per half-wave hold one row) ----
    float p[8];
#pragma unroll
    for (int r = 0; r < 8; ++r) {
        float x = s[r];
        float m = x;
        m = fmaxf(m, SWZ(m, 8));
        m = fmaxf(m, SWZ(m, 4));
        m = fmaxf(m, SWZ(m, 2));
        m = fmaxf(m, SWZ(m, 1));
        float e = __expf(x - m);
        float t = e;
        t += SWZ(t, 8);
        t += SWZ(t, 4);
        t += SWZ(t, 2);
        t += SWZ(t, 1);
        p[r] = e / t;
    }

    // ---- repack attn (C layout) -> A fragment via LDS, K=16..31 zero pad ----
#pragma unroll
    for (int r = 0; r < 8; ++r)
        attnS[wave][r + 8 * hi][lm] = f2bf(p[r]);
    __syncthreads();

    Frag af;
    af.u[0] = *(const uint4*)&attnS[wave][lm][hi * 8];
    af.u[1].x = 0u; af.u[1].y = 0u; af.u[1].z = 0u; af.u[1].w = 0u;

    // ---- ctx = attn(16x16,padK32) @ V(16x64): 4 WMMAs over d-chunks ----
#pragma unroll
    for (int nt = 0; nt < 4; ++nt) {
        Frag bfr;
        // B frag: lane = column d; elements 0..7 hold K = g = 8*hi..8*hi+7, elements 8..15 are K = 16..31 zero padding
#pragma unroll
        for (int g = 0; g < 16; ++g)
            bfr.h[g] = (g < 8) ? v[(hi * 8 + g) * HSZ + nt * 16 + lm] : (unsigned short)0;
        f32x8 c;
#pragma unroll
        for (int e = 0; e < 8; ++e) c[e] = 0.0f;
        c = __builtin_amdgcn_wmma_f32_16x16x32_bf16(
            false, af.v, false, bfr.v, (short)0, c, false, false);
#pragma unroll
        for (int r = 0; r < 8; ++r)
            Ctx[token * (size_t)DIMK + (size_t)(r + 8 * hi) * HSZ + nt * 16 + lm] = f2bf(c[r]);
    }
}

// ---------------------------------------------------------------------------
// Launcher
// ---------------------------------------------------------------------------
extern "C" void kernel_launch(void* const* d_in, const int* in_sizes, int n_in,
                              void* d_out, int out_size, void* d_ws, size_t ws_size,
                              hipStream_t stream)
{
    const float* x  = (const float*)d_in[0];
    const float* Wq = (const float*)d_in[1];
    const float* bq = (const float*)d_in[2];
    const float* Wk = (const float*)d_in[3];
    const float* bk = (const float*)d_in[4];
    const float* Wv = (const float*)d_in[5];
    const float* bv = (const float*)d_in[6];
    const float* Wo = (const float*)d_in[7];
    const float* bo = (const float*)d_in[8];

    const int D = DIMK;
    const long T = (long)in_sizes[0] / D;   // B*N tokens (32768)
    const long TD = T * (long)D;

    // workspace layout (bf16 / ushort elements)
    unsigned short* xb   = (unsigned short*)d_ws;
    unsigned short* qb   = xb   + TD;
    unsigned short* kb   = qb   + TD;
    unsigned short* vb   = kb   + TD;
    unsigned short* ctxb = vb   + TD;
    unsigned short* wqT  = ctxb + TD;
    unsigned short* wkT  = wqT + (size_t)D * D;
    unsigned short* wvT  = wkT + (size_t)D * D;
    unsigned short* woT  = wvT + (size_t)D * D;

    // 1) convert x to bf16
    {
        long blocks = (TD / 4 + 255) / 256;
        cvt_f32_to_bf16<<<dim3((unsigned)blocks), dim3(256), 0, stream>>>(x, xb, TD);
    }

    // 2) transpose+convert all four weights: WT[n][k] = bf16(W[k][n])
    {
        dim3 g(D / 64, D / 64);
        transpose_cvt<<<g, 256, 0, stream>>>(Wq, wqT, D);
        transpose_cvt<<<g, 256, 0, stream>>>(Wk, wkT, D);
        transpose_cvt<<<g, 256, 0, stream>>>(Wv, wvT, D);
        transpose_cvt<<<g, 256, 0, stream>>>(Wo, woT, D);
    }

    // 3) QKV projections: q/k/v = x @ W + b   (bf16 out)
    {
        dim3 g(D / BN, (unsigned)(T / BM));
        wmma_gemm<true><<<g, 256, 0, stream>>>(xb, wqT, bq, qb, (int)T, D, D);
        wmma_gemm<true><<<g, 256, 0, stream>>>(xb, wkT, bk, kb, (int)T, D, D);
        wmma_gemm<true><<<g, 256, 0, stream>>>(xb, wvT, bv, vb, (int)T, D, D);
    }

    // 4) per-token attention across heads
    {
        long blocks = T / 8;
        attention_kernel<<<dim3((unsigned)blocks), 256, 0, stream>>>(qb, kb, vb, ctxb);
    }

    // 5) output projection: out = ctx @ Wo + bo   (f32 out)
    {
        dim3 g(D / BN, (unsigned)(T / BM));
        wmma_gemm<false><<<g, 256, 0, stream>>>(ctxb, woT, bo, (float*)d_out, (int)T, D, D);
    }
}
